// DKMIL_41412074668554
// MI455X (gfx1250) — matrix operand order repaired and checked by reading, not compiled, not run on hardware
//
#include <hip/hip_runtime.h>
#include <hip/hip_bf16.h>

typedef __attribute__((ext_vector_type(16))) _Float16 v16h;
typedef __attribute__((ext_vector_type(8)))  float    v8f;

union Frag16 { unsigned int u[8]; v16h v; };

#define ACT_NONE  0
#define ACT_LRELU 1
#define ACT_TANH  2
#define ACT_DIST  3   // out = sqrt(max(xn[row]+kn[col]-2*acc, 0))

__device__ __forceinline__ float lrelu_f(float x) { return x > 0.f ? x : 0.01f * x; }

// 16 staged floats per thread (4x float4 -> global_load_b128)
struct R16 { float4 q[4]; };

// row-contiguous pattern: thread covers dst[row][kc..kc+16], row = t>>1, kc = (t&1)*16
__device__ __forceinline__ R16 load_rowpat(const float* __restrict__ src, int ld, int base,
                                           int t, int kb, int K)
{
    R16 r;
    const int row = t >> 1, kc = (t & 1) * 16;
    const int k0 = kb + kc;
    const float* p = src + (size_t)(base + row) * ld + k0;
    if (k0 + 16 <= K) {
        r.q[0] = *(const float4*)(p + 0);
        r.q[1] = *(const float4*)(p + 4);
        r.q[2] = *(const float4*)(p + 8);
        r.q[3] = *(const float4*)(p + 12);
    } else {
        float* f = (float*)&r;
        #pragma unroll
        for (int j = 0; j < 16; ++j) f[j] = (k0 + j < K) ? p[j] : 0.f;
    }
    return r;
}

__device__ __forceinline__ void store_rowpat(_Float16* __restrict__ dst, int t, const R16& r)
{
    const int row = t >> 1, kc = (t & 1) * 16;
    union { _Float16 h[16]; uint4 q[2]; } u;
    const float* f = (const float*)&r;
    #pragma unroll
    for (int j = 0; j < 16; ++j) u.h[j] = (_Float16)f[j];
    *(uint4*)&dst[row * 32 + kc + 0] = u.q[0];
    *(uint4*)&dst[row * 32 + kc + 8] = u.q[1];
}

// scatter pattern (transposing stage): thread covers dst[xc..xc+16][k], k = t>>3, xc = (t&7)*16
__device__ __forceinline__ R16 load_scatpat(const float* __restrict__ src, int ld, int base,
                                            int t, int kb, int K)
{
    R16 r;
    const int k = t >> 3, xc = (t & 7) * 16;
    if (kb + k < K) {
        const float* p = src + (size_t)(kb + k) * ld + base + xc;
        r.q[0] = *(const float4*)(p + 0);
        r.q[1] = *(const float4*)(p + 4);
        r.q[2] = *(const float4*)(p + 8);
        r.q[3] = *(const float4*)(p + 12);
    } else {
        float* f = (float*)&r;
        #pragma unroll
        for (int j = 0; j < 16; ++j) f[j] = 0.f;
    }
    return r;
}

__device__ __forceinline__ void store_scatpat(_Float16* __restrict__ dst, int t, const R16& r)
{
    const int k = t >> 3, xc = (t & 7) * 16;
    const float* f = (const float*)&r;
    #pragma unroll
    for (int j = 0; j < 16; ++j) dst[(xc + j) * 32 + k] = (_Float16)f[j];
}

// ---------------------------------------------------------------------------
// WMMA GEMM, 128x128 block tile, BK=32, ping-pong LDS software pipeline.
//   C[M,N] = act(A*B + bias) (+ resid)
//   TA=0: A[M,K] row-major   TA=1: A[K,M] row-major
//   TB=0: B[K,N] row-major   TB=1: B[N,K] row-major
// 8 waves as 4x2; each wave owns a 32x64 tile = 2x4 v_wmma_f32_16x16x32_f16.
// ---------------------------------------------------------------------------
template<int TA, int TB>
__global__ __launch_bounds__(256)
void gemm_wmma_kern(const float* __restrict__ A, const float* __restrict__ B,
                    const float* __restrict__ bias, const float* __restrict__ resid,
                    const float* __restrict__ xn, const float* __restrict__ kn,
                    float* __restrict__ C,
                    int M, int N, int K, int lda, int ldb, int ldc, int act)
{
    __shared__ __align__(16) _Float16 As[2][128 * 32];   // [m][k]
    __shared__ __align__(16) _Float16 Bs[2][128 * 32];   // [n][k]

    const int t    = threadIdx.x;
    const int wave = t >> 5;
    const int lane = t & 31;
    const int wr   = wave >> 1;        // 0..3 -> 32-row strip
    const int wc   = wave & 1;         // 0..1 -> 64-col strip
    const int lsub = lane & 15;
    const int h    = lane >> 4;
    const int m0   = blockIdx.y * 128;
    const int n0   = blockIdx.x * 128;

    const v8f vzero = {};
    v8f acc[2][4];
    #pragma unroll
    for (int i = 0; i < 2; ++i)
        #pragma unroll
        for (int j = 0; j < 4; ++j) acc[i][j] = vzero;

    auto loadA  = [&](int kb) { return TA ? load_scatpat(A, lda, m0, t, kb, K)
                                          : load_rowpat (A, lda, m0, t, kb, K); };
    auto storeA = [&](_Float16* d, const R16& r) { if (TA) store_scatpat(d, t, r);
                                                   else    store_rowpat (d, t, r); };
    auto loadB  = [&](int kb) { return TB ? load_rowpat (B, ldb, n0, t, kb, K)
                                          : load_scatpat(B, ldb, n0, t, kb, K); };
    auto storeB = [&](_Float16* d, const R16& r) { if (TB) store_rowpat (d, t, r);
                                                   else    store_scatpat(d, t, r); };

    // prologue: stage tile 0
    R16 ra = loadA(0);
    R16 rb = loadB(0);
    storeA(As[0], ra);
    storeB(Bs[0], rb);
    __syncthreads();

    const int nk = (K + 31) / 32;
    for (int it = 0; it < nk; ++it) {
        const int cur = it & 1, nxt = cur ^ 1;
        const bool has_next = (it + 1 < nk);
        if (has_next) {                       // global loads for next tile in flight
            ra = loadA((it + 1) * 32);
            rb = loadB((it + 1) * 32);
        }
        // compute current tile: 2 A-frags reused across 4 B-frags -> 8 WMMA
        Frag16 fa[2];
        #pragma unroll
        for (int i = 0; i < 2; ++i) {
            const unsigned int* ap = (const unsigned int*)&As[cur][(wr * 32 + i * 16 + lsub) * 32];
            #pragma unroll
            for (int p = 0; p < 8; ++p)
                fa[i].u[p] = ap[p + 4 * h + (p >= 4 ? 4 : 0)];   // A: k = 2p + 8h (+16)
        }
        #pragma unroll
        for (int j = 0; j < 4; ++j) {
            Frag16 fb;
            const unsigned int* bp = (const unsigned int*)&Bs[cur][(wc * 64 + j * 16 + lsub) * 32];
            #pragma unroll
            for (int p = 0; p < 8; ++p)
                fb.u[p] = bp[p + 4 * h + (p >= 4 ? 4 : 0)];       // B: k = 2p + 8h (+16)
            acc[0][j] = __builtin_amdgcn_wmma_f32_16x16x32_f16(false, fa[0].v, false, fb.v,
                                                               (short)0, acc[0][j], false, false);
            acc[1][j] = __builtin_amdgcn_wmma_f32_16x16x32_f16(false, fa[1].v, false, fb.v,
                                                               (short)0, acc[1][j], false, false);
        }
        if (has_next) {                       // commit next tile to the other buffer
            storeA(As[nxt], ra);
            storeB(Bs[nxt], rb);
        }
        __syncthreads();
    }

    // epilogue: lane holds col = lsub(+16j), rows r + 8h within each 16-row sub-tile
    #pragma unroll
    for (int i = 0; i < 2; ++i) {
        #pragma unroll
        for (int j = 0; j < 4; ++j) {
            const int col = n0 + wc * 64 + j * 16 + lsub;
            #pragma unroll
            for (int r = 0; r < 8; ++r) {
                const int row = m0 + wr * 32 + i * 16 + r + 8 * h;
                if (row < M && col < N) {
                    float v = acc[i][j][r];
                    if (bias) v += bias[col];
                    if (act == ACT_LRELU)      v = lrelu_f(v);
                    else if (act == ACT_TANH)  v = tanhf(v);
                    else if (act == ACT_DIST)  v = sqrtf(fmaxf(xn[row] + kn[col] - 2.f * v, 0.f));
                    if (resid) v += resid[(size_t)row * ldc + col];
                    C[(size_t)row * ldc + col] = v;
                }
            }
        }
    }
}

// ---------------------------------------------------------------------------
// Fused LeNet front-end: conv(1->20,k5)+lrelu+pool2, conv(20->50,k5)+relu+pool2
// One workgroup per instance; conv2 weights staged as f16 in LDS (~61KB total).
// ---------------------------------------------------------------------------
__global__ __launch_bounds__(256)
void dt_conv_kernel(const float* __restrict__ Bimg, const float* __restrict__ Kimg,
                    const float* __restrict__ c1w, const float* __restrict__ c1b,
                    const float* __restrict__ c2w, const float* __restrict__ c2b,
                    float* __restrict__ feat)
{
    __shared__ float    simg[784];
    __shared__ float    sw1[500];
    __shared__ _Float16 sp1[20 * 12 * 12];
    __shared__ _Float16 sw2[25000];

    const int inst = blockIdx.x;
    const int t = threadIdx.x;
    const float* img = (inst < 4096) ? (Bimg + (size_t)inst * 784)
                                     : (Kimg + (size_t)(inst - 4096) * 784);
    for (int i = t; i < 784;   i += 256) simg[i] = img[i];
    for (int i = t; i < 500;   i += 256) sw1[i]  = c1w[i];
    for (int i = t; i < 25000; i += 256) sw2[i]  = (_Float16)c2w[i];
    __syncthreads();

    for (int o = t; o < 2880; o += 256) {
        int c = o / 144, rem = o % 144, py = rem / 12, px = rem % 12;
        float best = -3.4e38f;
        for (int dy = 0; dy < 2; ++dy)
            for (int dx = 0; dx < 2; ++dx) {
                int oy = py * 2 + dy, ox = px * 2 + dx;
                float a = c1b[c];
                #pragma unroll
                for (int ky = 0; ky < 5; ++ky)
                    #pragma unroll
                    for (int kx = 0; kx < 5; ++kx)
                        a += simg[(oy + ky) * 28 + ox + kx] * sw1[c * 25 + ky * 5 + kx];
                best = fmaxf(best, lrelu_f(a));
            }
        sp1[o] = (_Float16)best;
    }
    __syncthreads();

    for (int o = t; o < 800; o += 256) {
        int c = o / 16, rem = o % 16, py = rem / 4, px = rem % 4;
        float best = -3.4e38f;
        for (int dy = 0; dy < 2; ++dy)
            for (int dx = 0; dx < 2; ++dx) {
                int oy = py * 2 + dy, ox = px * 2 + dx;
                float a = c2b[c];
                for (int ic = 0; ic < 20; ++ic) {
                    const _Float16* wr = &sw2[(c * 20 + ic) * 25];
                    const _Float16* pr = &sp1[ic * 144 + oy * 12 + ox];
                    #pragma unroll
                    for (int ky = 0; ky < 5; ++ky)
                        #pragma unroll
                        for (int kx = 0; kx < 5; ++kx)
                            a += (float)pr[ky * 12 + kx] * (float)wr[ky * 5 + kx];
                }
                best = fmaxf(best, fmaxf(a, 0.f));
            }
        feat[(size_t)inst * 800 + o] = best;
    }
}

// ---------------- small support kernels ----------------

__global__ void col_stats_kernel(const float* __restrict__ S, float* __restrict__ cmax,
                                 float* __restrict__ csum, int M, int N)
{
    int c = blockIdx.x * blockDim.x + threadIdx.x;
    if (c >= N) return;
    float mx = -3.4e38f, sum = 0.f;
    for (int r = 0; r < M; ++r) {
        float v = S[(size_t)r * N + c];
        float nm = fmaxf(mx, v);
        sum = sum * __expf(mx - nm) + __expf(v - nm);
        mx = nm;
    }
    cmax[c] = mx; csum[c] = sum;
}

__global__ void col_norm_kernel(float* __restrict__ S, const float* __restrict__ cmax,
                                const float* __restrict__ csum, int M, int N)
{
    size_t i = (size_t)blockIdx.x * blockDim.x + threadIdx.x;
    size_t total = (size_t)M * N;
    if (i >= total) return;
    int c = (int)(i % N);
    S[i] = __expf(S[i] - cmax[c]) / csum[c];
}

__global__ void rownorm2_kernel(const float* __restrict__ X, float* __restrict__ out,
                                int M, int K)
{
    int r = blockIdx.x * blockDim.x + threadIdx.x;
    if (r >= M) return;
    float s = 0.f;
    for (int k = 0; k < K; ++k) { float v = X[(size_t)r * K + k]; s += v * v; }
    out[r] = s;
}

__global__ void elem_mul_kernel(const float* __restrict__ a, const float* __restrict__ b,
                                float* __restrict__ o, size_t n)
{
    size_t i = (size_t)blockIdx.x * blockDim.x + threadIdx.x;
    if (i < n) o[i] = a[i] * b[i];
}

__global__ void matvec_lrelu_kernel(const float* __restrict__ X, const float* __restrict__ w,
                                    const float* __restrict__ b, float* __restrict__ out,
                                    int M, int K)
{
    int r = blockIdx.x * blockDim.x + threadIdx.x;
    if (r >= M) return;
    float s = b[0];
    for (int k = 0; k < K; ++k) s += X[(size_t)r * K + k] * w[k];
    out[r] = lrelu_f(s);
}

__global__ void mask_score_kernel(const float* __restrict__ T, const float* __restrict__ f4w,
                                  const float* __restrict__ f4b, float* __restrict__ m, int nbag)
{
    int j = blockIdx.x * blockDim.x + threadIdx.x;
    if (j >= nbag) return;
    float s = f4b[0];
    for (int i = 0; i < 128; ++i) s += T[(size_t)i * nbag + j] * f4w[i];
    m[j] = s;
}

__global__ void softmax1d_kernel(const float* __restrict__ in, float* __restrict__ out, int n)
{
    __shared__ float red[256];
    int t = threadIdx.x;
    float mx = -3.4e38f;
    for (int i = t; i < n; i += 256) mx = fmaxf(mx, in[i]);
    red[t] = mx; __syncthreads();
    for (int s = 128; s > 0; s >>= 1) { if (t < s) red[t] = fmaxf(red[t], red[t + s]); __syncthreads(); }
    mx = red[0]; __syncthreads();
    float sm = 0.f;
    for (int i = t; i < n; i += 256) sm += __expf(in[i] - mx);
    red[t] = sm; __syncthreads();
    for (int s = 128; s > 0; s >>= 1) { if (t < s) red[t] += red[t + s]; __syncthreads(); }
    sm = red[0]; __syncthreads();
    for (int i = t; i < n; i += 256) out[i] = __expf(in[i] - mx) / sm;
}

__global__ void topk_kernel(const float* __restrict__ m, float* __restrict__ scratch,
                            int* __restrict__ idx, int n, int kk)
{
    __shared__ float sv[256];
    __shared__ int   si[256];
    int t = threadIdx.x;
    for (int i = t; i < n; i += 256) scratch[i] = m[i];
    __syncthreads();
    for (int s = 0; s < kk; ++s) {
        float bv = -3.4e38f; int bi = n;
        for (int i = t; i < n; i += 256) {
            float v = scratch[i];
            if (v > bv || (v == bv && i < bi)) { bv = v; bi = i; }
        }
        sv[t] = bv; si[t] = bi; __syncthreads();
        for (int st = 128; st > 0; st >>= 1) {
            if (t < st) {
                if (sv[t + st] > sv[t] || (sv[t + st] == sv[t] && si[t + st] < si[t])) {
                    sv[t] = sv[t + st]; si[t] = si[t + st];
                }
            }
            __syncthreads();
        }
        if (t == 0) { idx[s] = si[0]; scratch[si[0]] = -3.4e38f; }
        __syncthreads();
    }
}

__global__ void build_fuse_kernel(const float* __restrict__ aff_s, const float* __restrict__ Bdt,
                                  const int* __restrict__ idx, float* __restrict__ fuse,
                                  int nbag, int nsamp)
{
    __shared__ float vals[204];
    __shared__ float red[256];
    int row = blockIdx.x, t = threadIdx.x;
    for (int j = t; j < nsamp; j += 256) vals[j] = aff_s[(size_t)row * nbag + idx[j]];
    __syncthreads();
    float mx = -3.4e38f;
    for (int j = t; j < nsamp; j += 256) mx = fmaxf(mx, vals[j]);
    red[t] = mx; __syncthreads();
    for (int s = 128; s > 0; s >>= 1) { if (t < s) red[t] = fmaxf(red[t], red[t + s]); __syncthreads(); }
    mx = red[0]; __syncthreads();
    float sm = 0.f;
    for (int j = t; j < nsamp; j += 256) sm += __expf(vals[j] - mx);
    red[t] = sm; __syncthreads();
    for (int s = 128; s > 0; s >>= 1) { if (t < s) red[t] += red[t + s]; __syncthreads(); }
    sm = red[0]; __syncthreads();
    const int ldf = 128 + nsamp;
    for (int c = t; c < 128; c += 256) fuse[(size_t)row * ldf + c] = Bdt[(size_t)row * 128 + c];
    for (int j = t; j < nsamp; j += 256)
        fuse[(size_t)row * ldf + 128 + j] = __expf(vals[j] - mx) / sm;
}

__global__ void weighted_colsum_kernel(const float* __restrict__ A, const float* __restrict__ H,
                                       float* __restrict__ out, int M, int cols)
{
    int c = blockIdx.x * blockDim.x + threadIdx.x;
    if (c >= cols) return;
    float s = 0.f;
    for (int r = 0; r < M; ++r) s += A[r] * H[(size_t)r * cols + c];
    out[c] = s;
}

__global__ void vecmat_lrelu_kernel(const float* __restrict__ v, const float* __restrict__ W,
                                    const float* __restrict__ b, float* __restrict__ out,
                                    int din, int dout)
{
    int j = blockIdx.x * blockDim.x + threadIdx.x;
    if (j >= dout) return;
    float s = b[j];
    for (int k = 0; k < din; ++k) s += v[k] * W[(size_t)k * dout + j];
    out[j] = lrelu_f(s);
}

__global__ void build_xc_kernel(const float* __restrict__ Bfuse, const float* __restrict__ Memb,
                                float* __restrict__ Xc, int n)
{
    size_t i = (size_t)blockIdx.x * blockDim.x + threadIdx.x;
    if (i >= (size_t)n * 256) return;
    int row = (int)(i / 256), c = (int)(i % 256);
    Xc[i] = (c < 128) ? Bfuse[(size_t)row * 128 + c] : Memb[c - 128];
}

__global__ void head_kernel(const float* __restrict__ AH2,
                            const float* __restrict__ emb2w, const float* __restrict__ emb2b,
                            const float* __restrict__ saw,   const float* __restrict__ sab,
                            const float* __restrict__ clsw,  const float* __restrict__ clsb,
                            float* __restrict__ out)
{
    __shared__ float b1[256];
    __shared__ float b2[128];
    int t = threadIdx.x;
    if (t < 256) {
        float s = emb2b[t];
        for (int k = 0; k < 256; ++k) s += AH2[k] * emb2w[(size_t)k * 256 + t];
        b1[t] = lrelu_f(s);
    }
    __syncthreads();
    if (t < 128) {
        float s = sab[t];
        for (int k = 0; k < 256; ++k) s += b1[k] * saw[(size_t)k * 128 + t];
        b2[t] = lrelu_f(s);
    }
    __syncthreads();
    if (t == 0) {
        float z = clsb[0];
        for (int k = 0; k < 128; ++k) z += b2[k] * clsw[k];
        float p = 1.f / (1.f + __expf(-z));
        p = fminf(fmaxf(p, 1e-5f), 1.f - 1e-5f);
        out[0] = p;
        out[1] = (p >= 0.5f) ? 1.f : 0.f;
    }
}

// ---------------------------------------------------------------------------
#define IN(i) ((const float*)d_in[(i)])

static inline void GEMM(hipStream_t st, const float* A, const float* B, const float* bias,
                        const float* resid, const float* xn, const float* kn, float* C,
                        int M, int N, int K, int lda, int ldb, int ldc,
                        int tA, int tB, int act)
{
    dim3 g((N + 127) / 128, (M + 127) / 128), b(256);
    if (!tA && !tB)
        gemm_wmma_kern<0,0><<<g, b, 0, st>>>(A, B, bias, resid, xn, kn, C, M, N, K, lda, ldb, ldc, act);
    else if (!tA && tB)
        gemm_wmma_kern<0,1><<<g, b, 0, st>>>(A, B, bias, resid, xn, kn, C, M, N, K, lda, ldb, ldc, act);
    else
        gemm_wmma_kern<1,0><<<g, b, 0, st>>>(A, B, bias, resid, xn, kn, C, M, N, K, lda, ldb, ldc, act);
}

extern "C" void kernel_launch(void* const* d_in, const int* in_sizes, int n_in,
                              void* d_out, int out_size, void* d_ws, size_t ws_size,
                              hipStream_t stream)
{
    (void)in_sizes; (void)n_in; (void)out_size; (void)ws_size;
    float* ws   = (float*)d_ws;
    float* outp = (float*)d_out;

    const size_t NB = 4096, NK = 2048, NT = NB + NK;
    // workspace layout (floats), with overlays noted
    const size_t F_FEAT = 0;                          // 6144*800  (later: fuse_in / B_fuse)
    const size_t F_DT   = F_FEAT + NT * 800;          // 6144*128  (B_dt | K_dt)
    const size_t F_X1   = F_DT   + NT * 128;          // 4096*128
    const size_t F_X2   = F_X1   + NB * 128;          // 4096*128
    const size_t F_X3   = F_X2   + NB * 128;          // 4096*2048
    const size_t F_S    = F_X3   + NB * 2048;         // 4096*4096 (later: H/Xc/H2/gates)
    const size_t F_XS   = F_S    + NB * NB;           // 4096*128
    const size_t F_KS   = F_XS   + NB * 128;          // 2048*128
    const size_t F_AFF  = F_KS   + NK * 128;          // 4096*2048 (aff, then aff_s in place)
    const size_t F_MS   = F_AFF  + NB * NK;
    const size_t F_CMAX = F_MS,            F_CSUM = F_CMAX + 4096;
    const size_t F_NX   = F_CSUM + 4096,   F_NK2  = F_NX + 4096;
    const size_t F_G    = F_NK2 + 2048,    F_T    = F_G + 128 * 128;
    const size_t F_M    = F_T + 128 * 2048, F_SCR = F_M + 2048;
    const size_t F_IDX  = F_SCR + 2048;   // int[204] stored here
    const size_t F_AL   = F_IDX + 256,     F_AS   = F_AL + 4096;
    const size_t F_AH   = F_AS + 4096,     F_ME   = F_AH + 256;
    const size_t F_AH2  = F_ME + 256,      F_A2L  = F_AH2 + 256;
    // overlays
    const size_t F_FUSE = F_FEAT;                     // 4096*332
    const size_t F_BF   = F_FEAT + NB * 332;          // 4096*128
    const size_t F_H    = F_S;                        // 4096*256
    const size_t F_XC   = F_H   + NB * 256;           // 4096*256
    const size_t F_H2   = F_XC  + NB * 256;           // 4096*256
    const size_t F_GV   = F_H2  + NB * 256;           // 4096*128
    const size_t F_GU   = F_GV  + NB * 128;
    const size_t F_G2V  = F_GU  + NB * 128;
    const size_t F_G2U  = F_G2V + NB * 128;

    // 1) fused conv front-end for B (0..4095) and K (4096..6143)
    dt_conv_kernel<<<(unsigned)NT, 256, 0, stream>>>(IN(0), IN(1), IN(2), IN(3), IN(4), IN(5),
                                                     ws + F_FEAT);
    // 2) embedding GEMM: dt = lrelu(feat800 @ e.w + e.b)
    GEMM(stream, ws + F_FEAT, IN(6), IN(7), nullptr, nullptr, nullptr, ws + F_DT,
         (int)NT, 128, 800, 800, 128, 128, 0, 0, ACT_LRELU);

    // skip_block helper
    auto skip_block = [&](const float* X, int n, int d,
                          const float* w1, const float* b1, const float* w2, const float* b2,
                          const float* w3, const float* b3, float* outP) {
        GEMM(stream, X, w1, b1, nullptr, nullptr, nullptr, ws + F_X1,
             n, 128, d, d, 128, 128, 0, 0, ACT_LRELU);
        GEMM(stream, X, w2, b2, nullptr, nullptr, nullptr, ws + F_X2,
             n, 128, d, d, 128, 128, 0, 0, ACT_LRELU);
        GEMM(stream, X, w3, b3, nullptr, nullptr, nullptr, ws + F_X3,
             n, d, d, d, d, d, 0, 0, ACT_LRELU);
        // S = X1 @ X2^T  [n,n]
        GEMM(stream, ws + F_X1, ws + F_X2, nullptr, nullptr, nullptr, nullptr, ws + F_S,
             n, n, 128, 128, 128, n, 0, 1, ACT_NONE);
        // softmax over axis 0 (columns), in place
        col_stats_kernel<<<(n + 255) / 256, 256, 0, stream>>>(ws + F_S, ws + F_CMAX, ws + F_CSUM, n, n);
        col_norm_kernel<<<(unsigned)(((size_t)n * n + 255) / 256), 256, 0, stream>>>(
            ws + F_S, ws + F_CMAX, ws + F_CSUM, n, n);
        // out = feat @ X3 + X
        GEMM(stream, ws + F_S, ws + F_X3, nullptr, X, nullptr, nullptr, outP,
             n, d, n, n, d, d, 0, 0, ACT_NONE);
    };

    // 3) skip blocks on embeddings
    skip_block(ws + F_DT,            (int)NB, 128, IN(8),  IN(9),  IN(10), IN(11), IN(12), IN(13), ws + F_XS);
    skip_block(ws + F_DT + NB * 128, (int)NK, 128, IN(14), IN(15), IN(16), IN(17), IN(18), IN(19), ws + F_KS);

    // 4) euclidean distances: aff = sqrt(max(|x|^2 + |k|^2 - 2 x.k, 0))
    rownorm2_kernel<<<(4096 + 255) / 256, 256, 0, stream>>>(ws + F_XS, ws + F_NX, (int)NB, 128);
    rownorm2_kernel<<<(2048 + 255) / 256, 256, 0, stream>>>(ws + F_KS, ws + F_NK2, (int)NK, 128);
    GEMM(stream, ws + F_XS, ws + F_KS, nullptr, nullptr, ws + F_NX, ws + F_NK2, ws + F_AFF,
         (int)NB, (int)NK, 128, 128, 128, (int)NK, 0, 1, ACT_DIST);

    // 5) skip block on affinity map (in place: residual==output buffer is element-safe)
    skip_block(ws + F_AFF, (int)NB, (int)NK, IN(20), IN(21), IN(22), IN(23), IN(24), IN(25), ws + F_AFF);

    // 6) mask block -> top-204 bag indices
    GEMM(stream, ws + F_AFF, IN(26), IN(27), nullptr, nullptr, nullptr, ws + F_X1,
         (int)NB, 128, (int)NK, (int)NK, 128, 128, 0, 0, ACT_LRELU);
    GEMM(stream, ws + F_AFF, IN(28), IN(29), nullptr, nullptr, nullptr, ws + F_X2,
         (int)NB, 128, (int)NK, (int)NK, 128, 128, 0, 0, ACT_TANH);
    // G = X1^T @ X2   [128,128], K=4096
    GEMM(stream, ws + F_X1, ws + F_X2, nullptr, nullptr, nullptr, nullptr, ws + F_G,
         128, 128, (int)NB, 128, 128, 128, 1, 0, ACT_NONE);
    // T = lrelu(G @ f3 + b)  [128,2048]
    GEMM(stream, ws + F_G, IN(30), IN(31), nullptr, nullptr, nullptr, ws + F_T,
         128, (int)NK, 128, 128, (int)NK, (int)NK, 0, 0, ACT_LRELU);
    mask_score_kernel<<<(2048 + 255) / 256, 256, 0, stream>>>(ws + F_T, IN(32), IN(33), ws + F_M, (int)NK);
    topk_kernel<<<1, 256, 0, stream>>>(ws + F_M, ws + F_SCR, (int*)(ws + F_IDX), (int)NK, 204);

    // 7) fuse_in = [B_dt | softmax(aff_s[:, idx])], B_fuse = lrelu(fuse_in @ kf + b)
    build_fuse_kernel<<<(unsigned)NB, 256, 0, stream>>>(ws + F_AFF, ws + F_DT, (int*)(ws + F_IDX),
                                                        ws + F_FUSE, (int)NK, 204);
    GEMM(stream, ws + F_FUSE, IN(34), IN(35), nullptr, nullptr, nullptr, ws + F_BF,
         (int)NB, 128, 332, 332, 128, 128, 0, 0, ACT_LRELU);

    // 8) attention block 1
    GEMM(stream, ws + F_BF, IN(36), IN(37), nullptr, nullptr, nullptr, ws + F_H,
         (int)NB, 256, 128, 128, 256, 256, 0, 0, ACT_LRELU);
    GEMM(stream, ws + F_H, IN(38), IN(39), nullptr, nullptr, nullptr, ws + F_GV,
         (int)NB, 128, 256, 256, 128, 128, 0, 0, ACT_TANH);
    GEMM(stream, ws + F_H, IN(40), IN(41), nullptr, nullptr, nullptr, ws + F_GU,
         (int)NB, 128, 256, 256, 128, 128, 0, 0, ACT_LRELU);
    elem_mul_kernel<<<(unsigned)((NB * 128 + 255) / 256), 256, 0, stream>>>(
        ws + F_GV, ws + F_GU, ws + F_GV, NB * 128);
    matvec_lrelu_kernel<<<(4096 + 255) / 256, 256, 0, stream>>>(ws + F_GV, IN(42), IN(43),
                                                                ws + F_AL, (int)NB, 128);
    softmax1d_kernel<<<1, 256, 0, stream>>>(ws + F_AL, ws + F_AS, (int)NB);
    weighted_colsum_kernel<<<1, 256, 0, stream>>>(ws + F_AS, ws + F_H, ws + F_AH, (int)NB, 256);
    vecmat_lrelu_kernel<<<1, 128, 0, stream>>>(ws + F_AH, IN(44), IN(45), ws + F_ME, 256, 128);

    // 9) Xc = [B_fuse | tile(M)]
    build_xc_kernel<<<(unsigned)((NB * 256 + 255) / 256), 256, 0, stream>>>(
        ws + F_BF, ws + F_ME, ws + F_XC, (int)NB);

    // 10) attention block 2 (A -> d_out[2:])
    GEMM(stream, ws + F_XC, IN(46), IN(47), nullptr, nullptr, nullptr, ws + F_H2,
         (int)NB, 256, 256, 256, 256, 256, 0, 0, ACT_LRELU);
    GEMM(stream, ws + F_H2, IN(48), IN(49), nullptr, nullptr, nullptr, ws + F_G2V,
         (int)NB, 128, 256, 256, 128, 128, 0, 0, ACT_TANH);
    GEMM(stream, ws + F_H2, IN(50), IN(51), nullptr, nullptr, nullptr, ws + F_G2U,
         (int)NB, 128, 256, 256, 128, 128, 0, 0, ACT_LRELU);
    elem_mul_kernel<<<(unsigned)((NB * 128 + 255) / 256), 256, 0, stream>>>(
        ws + F_G2V, ws + F_G2U, ws + F_G2V, NB * 128);
    matvec_lrelu_kernel<<<(4096 + 255) / 256, 256, 0, stream>>>(ws + F_G2V, IN(52), IN(53),
                                                                ws + F_A2L, (int)NB, 128);
    softmax1d_kernel<<<1, 256, 0, stream>>>(ws + F_A2L, outp + 2, (int)NB);
    weighted_colsum_kernel<<<1, 256, 0, stream>>>(outp + 2, ws + F_H2, ws + F_AH2, (int)NB, 256);

    // 11) classifier head -> d_out[0], d_out[1]
    head_kernel<<<1, 256, 0, stream>>>(ws + F_AH2, IN(54), IN(55), IN(56), IN(57), IN(58), IN(59), outp);
}
